// DeepOdoModel_44659069944183
// MI455X (gfx1250) — compile-verified, matrix operand order checked by reading, not run on hardware
//
#include <hip/hip_runtime.h>

typedef __attribute__((ext_vector_type(16))) _Float16 v16h;
typedef __attribute__((ext_vector_type(8)))  float    v8f;
typedef __attribute__((ext_vector_type(4)))  float    fx4;

__device__ __forceinline__ v16h ld_frag(const _Float16* p) {
  v16h r;
  reinterpret_cast<fx4*>(&r)[0] = *reinterpret_cast<const fx4*>(p);
  reinterpret_cast<fx4*>(&r)[1] = *reinterpret_cast<const fx4*>(p + 16);
  return r;
}

__device__ __forceinline__ v8f wmma16(v16h a, v16h b, v8f c) {
  return __builtin_amdgcn_wmma_f32_16x16x32_f16(false, a, false, b, (short)0, c,
                                                false, false);
}

// ---------------- f32 -> f16 weight conversion ----------------
__global__ void k_cvt(const float* __restrict__ src, _Float16* __restrict__ dst, int n) {
  int i = blockIdx.x * 256 + threadIdx.x;
  if (i < n) dst[i] = (_Float16)src[i];
}

// conv2 weights: [oc][ci][dt] f32  ->  [oc][k = dt*128 + ci] f16
// (K-permutation that makes the im2col B-columns contiguous)
__global__ void k_cvt_w2(const float* __restrict__ src, _Float16* __restrict__ dst) {
  int i = blockIdx.x * 256 + threadIdx.x;
  if (i < 256 * 1152) {
    int oc = i / 1152, r = i - oc * 1152;
    int dt = r >> 7, ci = r & 127;
    dst[i] = (_Float16)src[oc * 1152 + ci * 9 + dt];
  }
}

// ---------------- conv1 (7->128, k=11) + maxpool2 : VALU ----------------
// in : phone[n][l=50][c=7] ; out (TRANSPOSED): p1t[n][tp=20][oc=128] f16
__global__ __launch_bounds__(256) void k_conv1(const float* __restrict__ phone,
                                               const float* __restrict__ w1,
                                               const float* __restrict__ b1,
                                               _Float16* __restrict__ p1t) {
  __shared__ float xs[7 * 50];
  __shared__ float ws[128 * 77];
  int n = blockIdx.x, tid = threadIdx.x;
  const float* src = phone + (size_t)n * 350;
  for (int i = tid; i < 350; i += 256) {
    int l = i / 7, c = i - l * 7;
    xs[c * 50 + l] = src[i];
  }
  for (int i = tid; i < 128 * 77; i += 256) ws[i] = w1[i];  // [o][c][k] = o*77+c*11+k
  __syncthreads();
  _Float16* dst = p1t + (size_t)n * 2560;
  for (int idx = tid; idx < 2560; idx += 256) {
    int o = idx & 127, tp = idx >> 7;  // idx = tp*128 + o
    int t0 = tp * 2;
    float b = b1[o], a0 = b, a1 = b;
    const float* wo = ws + o * 77;
#pragma unroll
    for (int c = 0; c < 7; ++c)
#pragma unroll
      for (int k = 0; k < 11; ++k) {
        float w = wo[c * 11 + k];
        a0 += xs[c * 50 + t0 + k] * w;
        a1 += xs[c * 50 + t0 + 1 + k] * w;
      }
    dst[idx] = (_Float16)fmaxf(a0, a1);
  }
}

// ------- conv2 (128->256, k=9) + maxpool2 as sliding-window WMMA GEMM -------
// With p1t[n][tp][ci] and k = dt*128+ci, im2col column (n,t) is the contiguous
// slice p1t[n] + t*128 (length 1152). 4 frames/block -> 48 cols = 3 dense
// N-tiles; M = 256 (16 tiles, 2/wave); K = 1152 (36 k32 steps).
// All 5 fragment loads issued before the 6 WMMAs each K-step.
__global__ __launch_bounds__(256) void k_conv2(const _Float16* __restrict__ W2,
                                               const float* __restrict__ b2,
                                               const _Float16* __restrict__ p1t,
                                               _Float16* __restrict__ feat) {
  int tid = threadIdx.x;
  int w = tid >> 5, l = tid & 31, lr = l & 15, kh = l >> 4;
  int n0 = blockIdx.x * 4;
  int mt0 = w * 2;
  v8f acc[2][3] = {};
  int colt[3], coln[3];
  const _Float16* bp[3];
#pragma unroll
  for (int q = 0; q < 3; ++q) {
    int col = q * 16 + lr;
    int f = col / 12;
    colt[q] = col - f * 12;
    coln[q] = n0 + f;
    bp[q] = p1t + (size_t)(n0 + f) * 2560 + colt[q] * 128 + kh * 8;
  }
  const _Float16* a0p = W2 + (size_t)(mt0 * 16 + lr) * 1152 + kh * 8;
  const _Float16* a1p = a0p + (size_t)16 * 1152;
  for (int kt = 0; kt < 36; ++kt) {
    int k = kt * 32;
    v16h a0 = ld_frag(a0p + k);
    v16h a1 = ld_frag(a1p + k);
    v16h b0 = ld_frag(bp[0] + k);
    v16h b1 = ld_frag(bp[1] + k);
    v16h b2v = ld_frag(bp[2] + k);
    acc[0][0] = wmma16(a0, b0, acc[0][0]);
    acc[1][0] = wmma16(a1, b0, acc[1][0]);
    acc[0][1] = wmma16(a0, b1, acc[0][1]);
    acc[1][1] = wmma16(a1, b1, acc[1][1]);
    acc[0][2] = wmma16(a0, b2v, acc[0][2]);
    acc[1][2] = wmma16(a1, b2v, acc[1][2]);
  }
#pragma unroll
  for (int m = 0; m < 2; ++m) {
    int ocb = (mt0 + m) * 16 + kh * 8;  // row = v + 8*kh within tile
#pragma unroll
    for (int q = 0; q < 3; ++q) {
      int t = colt[q];
      int n = coln[q];
#pragma unroll
      for (int v = 0; v < 8; ++v) {
        int oc = ocb + v;
        float val = acc[m][q][v] + b2[oc];
        float other = __shfl_xor(val, 1, 32);  // partner column t^1, same frame
        if ((t & 1) == 0) {
          float mx = fmaxf(val, other);
          feat[(size_t)n * 1536 + oc * 6 + (t >> 1)] = (_Float16)mx;
        }
      }
    }
  }
}

// ---------------- generic WMMA GEMM: C[M,N] = A[M,K] * Bw[N,K]^T + bias ----------------
// block 256 thr = 8 waves; block tile 64x128; wave = 1 M-tile x 4 N-tiles.
// K multiple of 64; per K-step all 10 fragment loads issue before the 8 WMMAs.
__global__ __launch_bounds__(256) void k_gemm(const _Float16* __restrict__ A,
                                              const _Float16* __restrict__ Bw,
                                              const float* __restrict__ bias,
                                              void* __restrict__ out,
                                              int K, int N, int store_f32) {
  int tid = threadIdx.x;
  int w = tid >> 5, l = tid & 31, lr = l & 15, kh = l >> 4;
  int mt = w & 3, ng = (w >> 2) * 4;
  int row0 = blockIdx.x * 64 + mt * 16;
  const _Float16* ap = A + (size_t)(row0 + lr) * K + kh * 8;
  const _Float16* bp[4];
  int cols[4];
#pragma unroll
  for (int j = 0; j < 4; ++j) {
    cols[j] = blockIdx.y * 128 + (ng + j) * 16 + lr;
    bp[j] = Bw + (size_t)cols[j] * K + kh * 8;
  }
  v8f acc[4] = {};
  for (int k = 0; k < K; k += 64) {
    v16h a0 = ld_frag(ap + k);
    v16h a1 = ld_frag(ap + k + 32);
    v16h b[8];
#pragma unroll
    for (int j = 0; j < 4; ++j) {
      b[2 * j]     = ld_frag(bp[j] + k);
      b[2 * j + 1] = ld_frag(bp[j] + k + 32);
    }
#pragma unroll
    for (int j = 0; j < 4; ++j) acc[j] = wmma16(a0, b[2 * j], acc[j]);
#pragma unroll
    for (int j = 0; j < 4; ++j) acc[j] = wmma16(a1, b[2 * j + 1], acc[j]);
  }
#pragma unroll
  for (int j = 0; j < 4; ++j) {
    int c = cols[j];
    float bj = bias[c];
#pragma unroll
    for (int v = 0; v < 8; ++v) {
      int r = row0 + kh * 8 + v;
      float val = acc[j][v] + bj;
      size_t idx = (size_t)r * N + c;
      if (store_f32) ((float*)out)[idx] = val;
      else           ((_Float16*)out)[idx] = (_Float16)val;
    }
  }
}

// ---------------- sequential GRU: one workgroup, 32 waves ----------------
// h in LDS (f32 + f16 mirror). Per step: gh = h@Whh^T via WMMA (96 N-tiles,
// 3 per wave, K=512, A-frag loaded once per k32), gate math, fused fc3.
__global__ __launch_bounds__(1024) void k_gru(const float* __restrict__ h0,
                                              const _Float16* __restrict__ Whh,
                                              const float* __restrict__ GI,
                                              const float* __restrict__ bhh,
                                              const float* __restrict__ w3,
                                              const float* __restrict__ b3,
                                              float* __restrict__ ghbuf,
                                              float* __restrict__ out) {
  __shared__ float    hs[16 * 512];
  __shared__ _Float16 hh[16 * 512];
  __shared__ float    red[32];
  int tid = threadIdx.x;
  int w = tid >> 5, l = tid & 31, lr = l & 15, kh = l >> 4;
  for (int i = tid; i < 8192; i += 1024) {
    float v = h0[i];
    hs[i] = v;
    hh[i] = (_Float16)v;
  }
  __syncthreads();
  float fb3 = b3[0];
  int ncol[3];
  const _Float16* bp[3];
#pragma unroll
  for (int q = 0; q < 3; ++q) {
    ncol[q] = (w * 3 + q) * 16 + lr;
    bp[q] = Whh + (size_t)ncol[q] * 512 + kh * 8;
  }
  const _Float16* apv = hh + lr * 512 + kh * 8;
  for (int t = 0; t < 512; ++t) {
    // --- gh = h @ Whh^T  (M=16, N=1536, K=512) ---
    v8f acc[3] = {};
#pragma unroll 4
    for (int k = 0; k < 512; k += 32) {
      v16h a  = ld_frag(apv + k);
      v16h b0 = ld_frag(bp[0] + k);
      v16h b1 = ld_frag(bp[1] + k);
      v16h b2v = ld_frag(bp[2] + k);
      acc[0] = wmma16(a, b0, acc[0]);
      acc[1] = wmma16(a, b1, acc[1]);
      acc[2] = wmma16(a, b2v, acc[2]);
    }
#pragma unroll
    for (int q = 0; q < 3; ++q)
#pragma unroll
      for (int v = 0; v < 8; ++v) {
        int bb = kh * 8 + v;  // batch row
        ghbuf[bb * 1536 + ncol[q]] = acc[q][v];
      }
    __syncthreads();
    // --- gate activations + h update ---
    for (int i = tid; i < 8192; i += 1024) {
      int b = i >> 9, j = i & 511;
      size_t gbase = ((size_t)(b * 512 + t)) * 1536 + j;
      float gr = GI[gbase], gz = GI[gbase + 512], gn = GI[gbase + 1024];
      float hr = ghbuf[b * 1536 + j] + bhh[j];
      float hz = ghbuf[b * 1536 + 512 + j] + bhh[512 + j];
      float hn = ghbuf[b * 1536 + 1024 + j] + bhh[1024 + j];
      float r = 1.f / (1.f + __expf(-(gr + hr)));
      float z = 1.f / (1.f + __expf(-(gz + hz)));
      float nn = tanhf(gn + r * hn);
      float hnew = (1.f - z) * nn + z * hs[i];
      hs[i] = hnew;
      hh[i] = (_Float16)hnew;
    }
    __syncthreads();
    // --- fused fc3: out[b, t] = h_t[b,:] . w3 + b3 ---
    {
      int b = w & 15, jb = (w >> 4) * 256;
      float p = 0.f;
#pragma unroll
      for (int q = 0; q < 8; ++q) {
        int j = jb + l + 32 * q;
        p += hs[b * 512 + j] * w3[j];
      }
      for (int off = 16; off; off >>= 1) p += __shfl_down(p, off, 32);
      if (l == 0) red[w] = p;
    }
    __syncthreads();
    if (tid < 16) out[tid * 512 + t] = red[tid] + red[tid + 16] + fb3;
    __syncthreads();
  }
}

// ---------------- host side ----------------
extern "C" void kernel_launch(void* const* d_in, const int* in_sizes, int n_in,
                              void* d_out, int out_size, void* d_ws, size_t ws_size,
                              hipStream_t stream) {
  const float* phone = (const float*)d_in[0];
  const float* h0    = (const float*)d_in[1];
  const float* c1w   = (const float*)d_in[2];
  const float* c1b   = (const float*)d_in[3];
  const float* c2w   = (const float*)d_in[4];
  const float* c2b   = (const float*)d_in[5];
  const float* f1w   = (const float*)d_in[6];
  const float* f1b   = (const float*)d_in[7];
  const float* f2w   = (const float*)d_in[8];
  const float* f2b   = (const float*)d_in[9];
  const float* wih   = (const float*)d_in[10];
  const float* whh   = (const float*)d_in[11];
  const float* bih   = (const float*)d_in[12];
  const float* bhh   = (const float*)d_in[13];
  const float* w3    = (const float*)d_in[14];
  const float* b3    = (const float*)d_in[15];
  float* out = (float*)d_out;

  char* ws = (char*)d_ws;
  size_t off = 0;
  auto alloc = [&](size_t bytes) -> void* {
    void* p = ws + off;
    off += (bytes + 255) & ~(size_t)255;
    return p;
  };
  const int N = 8192;
  _Float16* W2h  = (_Float16*)alloc((size_t)256 * 1152 * 2);
  _Float16* F1h  = (_Float16*)alloc((size_t)1024 * 1536 * 2);
  _Float16* F2h  = (_Float16*)alloc((size_t)512 * 1024 * 2);
  _Float16* WIHh = (_Float16*)alloc((size_t)1536 * 512 * 2);
  _Float16* WHHh = (_Float16*)alloc((size_t)1536 * 512 * 2);
  _Float16* p1t  = (_Float16*)alloc((size_t)N * 2560 * 2);
  _Float16* feat = (_Float16*)alloc((size_t)N * 1536 * 2);
  _Float16* fc1o = (_Float16*)alloc((size_t)N * 1024 * 2);
  _Float16* xh   = (_Float16*)alloc((size_t)N * 512 * 2);
  float*    GI   = (float*)alloc((size_t)N * 1536 * 4);
  float*    ghb  = (float*)alloc((size_t)16 * 1536 * 4);

  auto cvt = [&](const float* s, _Float16* d, int n) {
    k_cvt<<<(n + 255) / 256, 256, 0, stream>>>(s, d, n);
  };
  k_cvt_w2<<<(256 * 1152 + 255) / 256, 256, 0, stream>>>(c2w, W2h);
  cvt(f1w, F1h, 1024 * 1536);
  cvt(f2w, F2h, 512 * 1024);
  cvt(wih, WIHh, 1536 * 512);
  cvt(whh, WHHh, 1536 * 512);

  k_conv1<<<N, 256, 0, stream>>>(phone, c1w, c1b, p1t);
  k_conv2<<<N / 4, 256, 0, stream>>>(W2h, c2b, p1t, feat);
  k_gemm<<<dim3(N / 64, 1024 / 128), 256, 0, stream>>>(feat, F1h, f1b, fc1o, 1536, 1024, 0);
  k_gemm<<<dim3(N / 64, 512 / 128), 256, 0, stream>>>(fc1o, F2h, f2b, xh, 1024, 512, 0);
  k_gemm<<<dim3(N / 64, 1536 / 128), 256, 0, stream>>>(xh, WIHh, bih, GI, 512, 1536, 1);
  k_gru<<<1, 1024, 0, stream>>>(h0, WHHh, GI, bhh, w3, b3, ghb, out);
}
